// HebbianMap2d_51041391345949
// MI455X (gfx1250) — compile-verified, matrix operand order checked by reading, not run on hardware
//
#include <hip/hip_runtime.h>
#include <hip/hip_bf16.h>

// ---------------------------------------------------------------------------
// HebbianMap2d on MI455X (gfx1250, wave32, WMMA)
//   x:      [16, 32, 64, 64]  f32
//   weight: [256, 32, 5, 5]   f32
//   out:    y_out [16,256,60,60] f32  ++  new_weight [256,32,5,5] f32
// Shapes: N = 16*60*60 = 57600 patches, D = 32*25 = 800, Cout = 256.
// Both big GEMMs run on v_wmma_f32_16x16x32_f16 (f16 in, f32 accumulate).
// ---------------------------------------------------------------------------

typedef __attribute__((ext_vector_type(16))) _Float16 v16h;
typedef __attribute__((ext_vector_type(8)))  _Float16 v8h;
typedef __attribute__((ext_vector_type(8)))  float    v8f;

#define N_PATCH   57600      // 16*60*60
#define D_DIM     800        // 32*5*5
#define COUT      256
#define Y_ELEMS   23040000   // 16*256*60*60
#define ETA       0.1f

// ---- WMMA helpers ---------------------------------------------------------

__device__ __forceinline__ v8f wmma_f16(v16h a, v16h b, v8f c) {
  // (neg_a, A, neg_b, B, c_mod, C, reuse_a, reuse_b)
  return __builtin_amdgcn_wmma_f32_16x16x32_f16(false, a, false, b,
                                                (short)0, c, false, false);
}

// Load one 16x32 f16 operand fragment from a row-major [16][32] LDS tile.
// 16-bit A layout (ISA 7.12.2): lane l -> row m = l&15, K-base = (l<16?0:8);
// VGPRs 0-3 hold K=base..base+7, VGPRs 4-7 hold K=base+16..base+23.
// Same lane pattern works for the B operand when the tile holds B^T rows.
__device__ __forceinline__ v16h ld_frag(const _Float16* tile, int lane) {
  int m  = lane & 15;
  int bk = (lane >> 4) << 3;                 // 0 or 8
  v8h lo = *(const v8h*)(tile + m * 32 + bk);
  v8h hi = *(const v8h*)(tile + m * 32 + bk + 16);
  return __builtin_shufflevector(lo, hi,
      0,1,2,3,4,5,6,7,8,9,10,11,12,13,14,15);
}

// ---- K0a: im2col x (f32) -> xu (f16), [N][D] row-major --------------------

__global__ __launch_bounds__(256) void k0_unfold(const float* __restrict__ x,
                                                 _Float16* __restrict__ xu) {
  int idx = blockIdx.x * 256 + threadIdx.x;          // 46,080,000 total
  if (idx >= N_PATCH * D_DIM) return;
  int n  = idx / D_DIM, d = idx - n * D_DIM;
  int b  = n / 3600,   rem = n - b * 3600;
  int hp = rem / 60,   wp  = rem - hp * 60;
  int ci = d / 25,     r2  = d - ci * 25;
  int i  = r2 / 5,     j   = r2 - i * 5;
  float v = x[(((long)(b * 32 + ci)) * 64 + (hp + i)) * 64 + (wp + j)];
  xu[idx] = (_Float16)v;
}

// ---- K0b: weight -> f16 + per-channel L2 norm -----------------------------

__global__ __launch_bounds__(256) void k0_wnorm(const float* __restrict__ w,
                                                _Float16* __restrict__ wh,
                                                float* __restrict__ normw) {
  __shared__ float red[256];
  int c = blockIdx.x, t = threadIdx.x;
  float acc = 0.f;
  for (int d = t; d < D_DIM; d += 256) {
    float v = w[c * D_DIM + d];
    wh[c * D_DIM + d] = (_Float16)v;
    acc += v * v;
  }
  red[t] = acc;
  __syncthreads();
  for (int s = 128; s > 0; s >>= 1) {
    if (t < s) red[t] += red[t + s];
    __syncthreads();
  }
  if (t == 0) {
    float nw = sqrtf(red[0]);
    if (nw == 0.f) nw = 1.f;
    normw[c] = nw;
  }
}

// ---- K1: GEMM1  proj[n,c] = sum_d xu[n,d]*wh[c,d];  y_out = proj/||w|| ----
// Block: 256 thr (8 waves), tile 128(n) x 128(c), K-step 32, 25 K-iters.
// Wave w: rows [ (w&3)*32, +32 ), cols [ (w>>2)*64, +64 ) -> 2x4 subtiles.

__global__ __launch_bounds__(256) void k1_gemm(
    const _Float16* __restrict__ xu, const _Float16* __restrict__ wh,
    const float* __restrict__ normw,
    float* __restrict__ yout, float* __restrict__ proj) {
  __shared__ union {
    struct { _Float16 A[128 * 32]; _Float16 B[128 * 32]; } ab;
    float pt[128 * 129];                    // epilogue transpose buffer
  } sm;

  int t = threadIdx.x, lane = t & 31, wv = t >> 5;
  int n0 = blockIdx.x * 128;                // 450 blocks
  int c0 = blockIdx.y * 128;                // 2 blocks
  int wm = wv & 3, wn = wv >> 2;

  v8f zero = {0.f,0.f,0.f,0.f,0.f,0.f,0.f,0.f};
  v8f acc[2][4];
  #pragma unroll
  for (int i = 0; i < 2; i++)
    #pragma unroll
    for (int j = 0; j < 4; j++) acc[i][j] = zero;

  for (int kk = 0; kk < D_DIM; kk += 32) {
    __syncthreads();
    #pragma unroll
    for (int i = 0; i < 2; i++) {           // 512 chunks of 8 f16 each
      int ch = t + i * 256;
      int r  = ch >> 2, c8 = (ch & 3) << 3;
      *(v8h*)&sm.ab.A[r * 32 + c8] =
          *(const v8h*)&xu[(long)(n0 + r) * D_DIM + kk + c8];
      *(v8h*)&sm.ab.B[r * 32 + c8] =
          *(const v8h*)&wh[(long)(c0 + r) * D_DIM + kk + c8];
    }
    __syncthreads();
    v16h afr[2], bfr[4];
    #pragma unroll
    for (int mt = 0; mt < 2; mt++)
      afr[mt] = ld_frag(&sm.ab.A[(wm * 32 + mt * 16) * 32], lane);
    #pragma unroll
    for (int nt = 0; nt < 4; nt++)
      bfr[nt] = ld_frag(&sm.ab.B[(wn * 64 + nt * 16) * 32], lane);
    #pragma unroll
    for (int mt = 0; mt < 2; mt++)
      #pragma unroll
      for (int nt = 0; nt < 4; nt++)
        acc[mt][nt] = wmma_f16(afr[mt], bfr[nt], acc[mt][nt]);
  }

  __syncthreads();                           // A/B reads done; reuse as pt
  // C/D layout: VGPR i -> M = i + 8*(lane>=16), N-col = lane&15
  #pragma unroll
  for (int mt = 0; mt < 2; mt++)
    #pragma unroll
    for (int nt = 0; nt < 4; nt++)
      #pragma unroll
      for (int i = 0; i < 8; i++) {
        int row = wm * 32 + mt * 16 + i + ((lane >> 4) << 3);
        int col = wn * 64 + nt * 16 + (lane & 15);
        sm.pt[row * 129 + col] = acc[mt][nt][i];
      }
  __syncthreads();

  // proj[N][256] row-major (coalesced over c)
  for (int idx = t; idx < 16384; idx += 256) {
    int r = idx >> 7, c = idx & 127;
    proj[(long)(n0 + r) * COUT + c0 + c] = sm.pt[r * 129 + c];
  }
  // y_out NCHW (coalesced over n -> contiguous hp*60+wp runs)
  for (int idx = t; idx < 16384; idx += 256) {
    int c = idx >> 7, r = idx & 127;
    int n = n0 + r;
    int b = n / 3600, rem = n - b * 3600;
    yout[(long)(b * COUT + c0 + c) * 3600 + rem] =
        sm.pt[r * 129 + c] / normw[c0 + c];
  }
}

// ---- K2: per-patch winner-take-all + Gaussian LFB -------------------------
// One wave per row, 4 rows per wave, 8 waves/block -> 32 rows/block, 1800 blks.

__global__ __launch_bounds__(256) void k2_winner(
    const float* __restrict__ proj, const _Float16* __restrict__ xu,
    _Float16* __restrict__ s_out, float* __restrict__ rpart) {
  __shared__ float lred[4096];               // [2][8 waves][256 ch]
  int t = threadIdx.x, wv = t >> 5, lane = t & 31;

  float racc[8], r2acc[8];
  #pragma unroll
  for (int s = 0; s < 8; s++) { racc[s] = 0.f; r2acc[s] = 0.f; }

  int rowbase = blockIdx.x * 32 + wv * 4;
  for (int rr = 0; rr < 4; rr++) {
    int n = rowbase + rr;
    // ||x_unf[n]||
    const _Float16* xr = xu + (long)n * D_DIM;
    float nx = 0.f;
    #pragma unroll
    for (int k = 0; k < 25; k++) {
      float v = (float)xr[lane + 32 * k];
      nx += v * v;
    }
    #pragma unroll
    for (int off = 16; off > 0; off >>= 1) nx += __shfl_xor(nx, off, 32);
    nx = sqrtf(nx);
    if (nx == 0.f) nx = 1.f;

    // sim = ((cos+1)/2)^2, wave-wide max
    const float* pr = proj + (long)n * COUT;
    float sim[8], smax = -1e30f;
    #pragma unroll
    for (int s = 0; s < 8; s++) {
      float cs = pr[lane + 32 * s] / nx;
      float u  = (cs + 1.f) * 0.5f;
      sim[s] = u * u;
      smax = fmaxf(smax, sim[s]);
    }
    #pragma unroll
    for (int off = 16; off > 0; off >>= 1)
      smax = fmaxf(smax, __shfl_xor(smax, off, 32));

    unsigned bal[8];
    #pragma unroll
    for (int s = 0; s < 8; s++)
      bal[s] = __builtin_amdgcn_ballot_w32(sim[s] == smax);

    // LFB: r[o] = sum over winners w of exp(-cheb(w,o)^2/98), |d|-window [-7,8]
    #pragma unroll
    for (int s = 0; s < 8; s++) {
      int oc = lane + 32 * s, oy = oc >> 4, ox = oc & 15;
      float r = 0.f;
      for (int s2 = 0; s2 < 8; s2++) {
        unsigned bm = bal[s2];
        while (bm) {
          int l = __ffs(bm) - 1;
          bm &= bm - 1;
          int wc = l + 32 * s2, wy = wc >> 4, wx = wc & 15;
          int dy = wy - oy, dx = wx - ox;
          if (dy >= -7 && dy <= 8 && dx >= -7 && dx <= 8) {
            int ay = dy < 0 ? -dy : dy, ax = dx < 0 ? -dx : dx;
            int ch = ay > ax ? ay : ax;
            r += __expf(-(float)(ch * ch) * (1.0f / 98.0f));
          }
        }
      }
      r = fminf(fmaxf(r, -1.f), 1.f);        // clip; r >= 0 here
      float r2 = r * r;                      // == r*|r| and |r|^2
      s_out[(long)n * COUT + oc] = (_Float16)r2;
      racc[s]  += r;                         // |r|
      r2acc[s] += r2;
    }
  }

  #pragma unroll
  for (int s = 0; s < 8; s++) {
    int oc = lane + 32 * s;
    lred[wv * 256 + oc]        = racc[s];
    lred[2048 + wv * 256 + oc] = r2acc[s];
  }
  __syncthreads();
  float sr = 0.f, sr2 = 0.f;
  #pragma unroll
  for (int w = 0; w < 8; w++) {
    sr  += lred[w * 256 + t];
    sr2 += lred[2048 + w * 256 + t];
  }
  rpart[(long)blockIdx.x * 512 + t]       = sr;
  rpart[(long)blockIdx.x * 512 + 256 + t] = sr2;
}

// ---- K3: reduce per-block channel partials --------------------------------

__global__ __launch_bounds__(256) void k3_reduce(const float* __restrict__ rpart,
                                                 float* __restrict__ rsum) {
  int c = threadIdx.x;
  float sr = 0.f, sr2 = 0.f;
  for (int b = 0; b < 1800; b++) {
    sr  += rpart[b * 512 + c];
    sr2 += rpart[b * 512 + 256 + c];
  }
  if (sr == 0.f) sr = 1.f;
  rsum[c]       = sr;                        // sum |r|
  rsum[256 + c] = sr2;                       // sum r^2
}

// ---- K4: GEMM2  num[c,d] = sum_n r2[n,c] * xu[n,d], split-K = 8 -----------
// Block: 256 thr (8 waves), tile 128(c) x 80(d), K chunk 7200, K-step 32.
// Wave w owns c-subtile w (16 rows) x 5 d-subtiles.

__global__ __launch_bounds__(256) void k4_gemm2(
    const _Float16* __restrict__ s_in, const _Float16* __restrict__ xu,
    float* __restrict__ gpart) {
  __shared__ _Float16 As[128 * 32];          // [c][k]
  __shared__ _Float16 Bs[80 * 32];           // [d][k]  (B^T rows)
  int t = threadIdx.x, lane = t & 31, wv = t >> 5;

  int bid = blockIdx.x;                      // 160 blocks
  int ks  = bid / 20;
  int rem = bid - ks * 20;
  int mb  = rem / 10, nb = rem - mb * 10;
  int c0  = mb * 128, d0 = nb * 80;
  long kbase = (long)ks * 7200;

  v8f zero = {0.f,0.f,0.f,0.f,0.f,0.f,0.f,0.f};
  v8f acc[5];
  #pragma unroll
  for (int i = 0; i < 5; i++) acc[i] = zero;

  for (int kk = 0; kk < 7200; kk += 32) {
    __syncthreads();
    // A: s_in[n][c] -> As[c][k]  (32 k-rows x 128 c, transpose on store)
    #pragma unroll
    for (int i = 0; i < 2; i++) {
      int ch = t + i * 256;                  // 512 chunks of 8 c
      int k  = ch >> 4, cc = (ch & 15) << 3;
      v8h v = *(const v8h*)&s_in[(kbase + kk + k) * COUT + c0 + cc];
      #pragma unroll
      for (int j = 0; j < 8; j++) As[(cc + j) * 32 + k] = v[j];
    }
    // B: xu[n][d] -> Bs[d][k]  (32 k-rows x 80 d)
    for (int ch = t; ch < 320; ch += 256) {
      int k  = ch / 10, dd = (ch - k * 10) << 3;
      v8h v = *(const v8h*)&xu[(kbase + kk + k) * D_DIM + d0 + dd];
      #pragma unroll
      for (int j = 0; j < 8; j++) Bs[(dd + j) * 32 + k] = v[j];
    }
    __syncthreads();
    v16h afr = ld_frag(&As[(wv * 16) * 32], lane);
    #pragma unroll
    for (int nt = 0; nt < 5; nt++) {
      v16h bfr = ld_frag(&Bs[(nt * 16) * 32], lane);
      acc[nt] = wmma_f16(afr, bfr, acc[nt]);
    }
  }

  #pragma unroll
  for (int nt = 0; nt < 5; nt++)
    #pragma unroll
    for (int i = 0; i < 8; i++) {
      int c = c0 + wv * 16 + i + ((lane >> 4) << 3);
      int d = d0 + nt * 16 + (lane & 15);
      gpart[((long)ks * COUT + c) * D_DIM + d] = acc[nt][i];
    }
}

// ---- K5: combine split-K + Hebbian weight update --------------------------

__global__ __launch_bounds__(256) void k5_update(
    const float* __restrict__ gpart, const float* __restrict__ rsum,
    const float* __restrict__ w_in, float* __restrict__ w_out) {
  int idx = blockIdx.x * 256 + threadIdx.x;  // 204,800 elements
  if (idx >= COUT * D_DIM) return;
  int c = idx / D_DIM;
  float a = 0.f;
  #pragma unroll
  for (int ks = 0; ks < 8; ks++) a += gpart[ks * (COUT * D_DIM) + idx];
  float wf  = w_in[idx];
  float num = a - rsum[256 + c] * wf;        // - (sum r^2) * w
  w_out[idx] = wf + ETA * num / rsum[c];
}

// ---------------------------------------------------------------------------

extern "C" void kernel_launch(void* const* d_in, const int* in_sizes, int n_in,
                              void* d_out, int out_size, void* d_ws, size_t ws_size,
                              hipStream_t stream) {
  (void)in_sizes; (void)n_in; (void)out_size; (void)ws_size;
  const float* x = (const float*)d_in[0];    // [16,32,64,64]
  const float* w = (const float*)d_in[1];    // [256,32,5,5] == [256,800]
  float* yout = (float*)d_out;               // [16,256,60,60]
  float* wout = yout + Y_ELEMS;              // [256,800]

  char* ws = (char*)d_ws;                    // ~191.3 MB, all 16B-aligned
  _Float16* xu    = (_Float16*)(ws);                     //  92,160,000 B
  _Float16* wh    = (_Float16*)(ws +  92160000);         //     409,600 B
  float*    normw = (float*)   (ws +  92569600);         //       1,024 B
  float*    proj  = (float*)   (ws +  92570624);         //  58,982,400 B
  _Float16* sv    = (_Float16*)(ws + 151553024);         //  29,491,200 B
  float*    rpart = (float*)   (ws + 181044224);         //   3,686,400 B
  float*    rsum  = (float*)   (ws + 184730624);         //       2,048 B
  float*    gpart = (float*)   (ws + 184732672);         //   6,553,600 B

  k0_unfold<<<180000, 256, 0, stream>>>(x, xu);
  k0_wnorm <<<COUT,   256, 0, stream>>>(w, wh, normw);
  k1_gemm  <<<dim3(450, 2), 256, 0, stream>>>(xu, wh, normw, yout, proj);
  k2_winner<<<1800,   256, 0, stream>>>(proj, xu, sv, rpart);
  k3_reduce<<<1,      256, 0, stream>>>(rpart, rsum);
  k4_gemm2 <<<160,    256, 0, stream>>>(sv, xu, gpart);
  k5_update<<<800,    256, 0, stream>>>(gpart, rsum, w, wout);
}
